// InCTRL_69922067579267
// MI455X (gfx1250) — matrix operand order repaired and checked by reading, not compiled, not run on hardware
//
#include <hip/hip_runtime.h>

typedef __attribute__((ext_vector_type(16))) __bf16 bf16x16;
typedef __attribute__((ext_vector_type(8)))  float  f32x8;

#define BNS 0.9999950000374997f   // 1/sqrt(1+1e-5)
#define AST 648                   // padded bf16 K-stride (conflict-free LDS)
#define MT  96                    // M rows per block (14400 = 150*96)

__device__ __forceinline__ unsigned short f2bf(float f) {
  union { float f; unsigned int u; } c; c.f = f;
  unsigned int r = c.u + 0x7FFFu + ((c.u >> 16) & 1u);   // RTNE
  return (unsigned short)(r >> 16);
}
__device__ __forceinline__ unsigned int pack2bf(float a, float b) {
  return (unsigned int)f2bf(a) | ((unsigned int)f2bf(b) << 16);
}

// ---------------- kernel 0: f32 -> bf16 for patch_embeds_ref -----------------
__global__ __launch_bounds__(256)
void cvt_bf16_kernel(const float* __restrict__ src, unsigned short* __restrict__ dst, int n) {
  int i = (blockIdx.x * 256 + threadIdx.x) * 4;
  if (i + 3 < n) {
    float4 f = *(const float4*)(src + i);
    uint2 u; u.x = pack2bf(f.x, f.y); u.y = pack2bf(f.z, f.w);
    *(uint2*)(dst + i) = u;
  }
}

// ---------------- kernel 1: fused normalize + GEMM + max_r(sim) --------------
// pe: [3,14400,640] f32 (un-normalized rows), refs: [3,1800,640] bf16 (unit rows)
// maxsim[l*14400+m] = max_r dot(pe_hat[l,m], refs[l,r])
__global__ __launch_bounds__(192)
void simmax_kernel(const float* __restrict__ pe, const unsigned short* __restrict__ refs,
                   float* __restrict__ maxsim) {
  __shared__ unsigned short As[MT * AST];   // 124416 B
  __shared__ unsigned short Bs[16 * AST];   //  20736 B
  const int tid  = threadIdx.x;
  const int lane = tid & 31;
  const int wv   = tid >> 5;        // 0..5, each wave owns 16 M rows
  const int half = lane >> 4;
  const int ln16 = lane & 15;
  const int l     = blockIdx.y;
  const int mbase = blockIdx.x * MT;

  // stage + L2-normalize this wave's 16 A rows, convert to bf16
  const float* abase = pe + ((size_t)(l * 14400 + mbase)) * 640;
  for (int rr = 0; rr < 16; ++rr) {
    const int r = wv * 16 + rr;
    const float* row = abase + (size_t)r * 640 + lane * 20;
    float v[20]; float ss = 0.f;
#pragma unroll
    for (int q = 0; q < 5; ++q) {
      float4 t = *(const float4*)(row + q * 4);
      v[q*4+0]=t.x; v[q*4+1]=t.y; v[q*4+2]=t.z; v[q*4+3]=t.w;
      ss += t.x*t.x + t.y*t.y + t.z*t.z + t.w*t.w;
    }
#pragma unroll
    for (int m = 16; m >= 1; m >>= 1) ss += __shfl_xor(ss, m, 32);
    const float inv = rsqrtf(ss);
    unsigned short* dst = &As[r * AST + lane * 20];
#pragma unroll
    for (int q = 0; q < 10; ++q)
      *(unsigned int*)(dst + 2 * q) = pack2bf(v[2*q] * inv, v[2*q+1] * inv);
  }

  float runmax[8];
#pragma unroll
  for (int r = 0; r < 8; ++r) runmax[r] = -3.0e38f;

  // A-frag (this code's contiguous k order, same as B): lane(m=ln16, half) holds K = kc+16*half+{0..15} contiguous
  const unsigned short* Abase0 = &As[(wv * 16 + ln16) * AST + 16 * half];
  // B-frag (this code's contiguous k order, same as A): lane(n=ln16, half) holds K = kc+16*half+{0..15} contiguous
  const unsigned short* Bbase0 = &Bs[ln16 * AST + 16 * half];

  // register double-buffer for the 16-col B tile (16*640 bf16 = 1280 x 16B chunks)
  uint4 pref[7];
  const unsigned short* refl = refs + (size_t)l * 1800 * 640;

  // prologue: prefetch tile 0
  {
#pragma unroll
    for (int i = 0; i < 7; ++i) {
      const int c = tid + i * 192;
      if (c < 1280) {
        const int col = c / 80, k8 = c - col * 80;
        if (col < 1800)
          pref[i] = *(const uint4*)(refl + (size_t)col * 640 + k8 * 8);
      }
    }
  }

  for (int nt = 0; nt < 113; ++nt) {               // 1800 refs in tiles of 16
    __syncthreads();                               // Bs no longer being read
    // store prefetched tile nt into LDS
#pragma unroll
    for (int i = 0; i < 7; ++i) {
      const int c = tid + i * 192;
      if (c < 1280) {
        const int col = c / 80, k8 = c - col * 80;
        if (nt * 16 + col < 1800)
          *(uint4*)(&Bs[col * AST + k8 * 8]) = pref[i];
      }
    }
    __syncthreads();                               // Bs ready

    // issue global prefetch of tile nt+1; overlaps with the WMMA chain below
    if (nt + 1 < 113) {
      const int ncol0 = (nt + 1) * 16;
#pragma unroll
      for (int i = 0; i < 7; ++i) {
        const int c = tid + i * 192;
        if (c < 1280) {
          const int col = c / 80, k8 = c - col * 80;
          const int n = ncol0 + col;
          if (n < 1800)
            pref[i] = *(const uint4*)(refl + (size_t)n * 640 + k8 * 8);
        }
      }
    }

    // two interleaved accumulation chains over K = 640 = 20 x 32
    f32x8 acc0 = {0.f,0.f,0.f,0.f,0.f,0.f,0.f,0.f};
    f32x8 acc1 = {0.f,0.f,0.f,0.f,0.f,0.f,0.f,0.f};
#pragma unroll
    for (int kk = 0; kk < 10; ++kk) {
      {
        const int kc = (2 * kk) * 32;
        union { bf16x16 v; uint4 q[2]; } a, b;
        a.q[0] = *(const uint4*)(Abase0 + kc);
        a.q[1] = *(const uint4*)(Abase0 + kc + 8);
        b.q[0] = *(const uint4*)(Bbase0 + kc);
        b.q[1] = *(const uint4*)(Bbase0 + kc + 8);
        acc0 = __builtin_amdgcn_wmma_f32_16x16x32_bf16(false, a.v, false, b.v,
                                                       (short)0, acc0, false, false);
      }
      {
        const int kc = (2 * kk + 1) * 32;
        union { bf16x16 v; uint4 q[2]; } a, b;
        a.q[0] = *(const uint4*)(Abase0 + kc);
        a.q[1] = *(const uint4*)(Abase0 + kc + 8);
        b.q[0] = *(const uint4*)(Bbase0 + kc);
        b.q[1] = *(const uint4*)(Bbase0 + kc + 8);
        acc1 = __builtin_amdgcn_wmma_f32_16x16x32_bf16(false, a.v, false, b.v,
                                                       (short)0, acc1, false, false);
      }
    }
    const f32x8 acc = acc0 + acc1;

    const int n = nt * 16 + ln16;
    if (n < 1800) {
#pragma unroll
      for (int r = 0; r < 8; ++r) runmax[r] = fmaxf(runmax[r], acc[r]);
    }
  }

  // reduce max over N across the 16 lanes of each half (C: VGPR r -> M=r+8*half)
#pragma unroll
  for (int r = 0; r < 8; ++r) {
    float v = runmax[r];
    v = fmaxf(v, __shfl_xor(v, 1, 32));
    v = fmaxf(v, __shfl_xor(v, 2, 32));
    v = fmaxf(v, __shfl_xor(v, 4, 32));
    v = fmaxf(v, __shfl_xor(v, 8, 32));
    runmax[r] = v;
  }
  if (ln16 == 0) {
    const int row0 = mbase + wv * 16 + half * 8;
#pragma unroll
    for (int r = 0; r < 8; ++r)
      maxsim[(size_t)l * 14400 + row0 + r] = runmax[r];
  }
}

// ---------------- kernel 2: per-batch text/adapter/ref-head + prm/fg ---------
__global__ __launch_bounds__(256)
void score_kernel(const float* __restrict__ x_all, const float* __restrict__ te,
                  const float* __restrict__ xref,
                  const float* __restrict__ a_w1, const float* __restrict__ a_w2,
                  const float* __restrict__ r_w1, const float* __restrict__ r_b1,
                  const float* __restrict__ r_g2, const float* __restrict__ r_be2,
                  const float* __restrict__ r_w2, const float* __restrict__ r_b2,
                  const float* __restrict__ r_g3, const float* __restrict__ r_be3,
                  const float* __restrict__ r_w3, const float* __restrict__ r_b3,
                  const float* __restrict__ maxsim,
                  float* __restrict__ prm, float* __restrict__ ts_o,
                  float* __restrict__ irs_o, float* __restrict__ fg_o) {
  const int b = blockIdx.x, tid = threadIdx.x, lane = tid & 31, wv = tid >> 5;
  __shared__ float xs[640], diffs[640], h1s[160], z1s[128], z2s[64];
  __shared__ float redA[8], redB[8], redC[8];

  for (int d = tid; d < 640; d += 256) xs[d] = x_all[b * 640 + d];
  __syncthreads();

  float ps = 0.f, p0 = 0.f, p1 = 0.f;
  for (int d = tid; d < 640; d += 256) {
    float xv = xs[d];
    ps += xv * xv; p0 += xv * te[d]; p1 += xv * te[640 + d];
  }
#pragma unroll
  for (int m = 16; m >= 1; m >>= 1) {
    ps += __shfl_xor(ps, m, 32);
    p0 += __shfl_xor(p0, m, 32);
    p1 += __shfl_xor(p1, m, 32);
  }
  if (lane == 0) { redA[wv] = ps; redB[wv] = p0; redC[wv] = p1; }
  __syncthreads();
  if (tid == 0) {
    float s = 0.f, a = 0.f, c = 0.f;
    for (int i = 0; i < 8; ++i) { s += redA[i]; a += redB[i]; c += redC[i]; }
    float invn = rsqrtf(s);
    ts_o[b] = 1.f / (1.f + expf(100.f * invn * (a - c)));   // softmax[:,1]
  }

  if (tid < 160) {                                           // adapter layer 1
    float acc = 0.f; const float* w = a_w1 + tid * 640;
    for (int d = 0; d < 640; ++d) acc += xs[d] * w[d];
    h1s[tid] = fmaxf(acc, 0.f);
  }
  __syncthreads();
  for (int d = tid; d < 640; d += 256) {                     // adapter layer 2 + diff
    float acc = 0.f; const float* w = a_w2 + d * 160;
    for (int j = 0; j < 160; ++j) acc += h1s[j] * w[j];
    diffs[d] = xref[d] - fmaxf(acc, 0.f);
  }
  __syncthreads();
  if (tid < 128) {                                           // r-head layer 1
    float acc = r_b1[tid]; const float* w = r_w1 + tid * 640;
    for (int d = 0; d < 640; ++d) acc += diffs[d] * w[d];
    z1s[tid] = fmaxf(acc, 0.f) * (r_g2[tid] * BNS) + r_be2[tid];
  }
  __syncthreads();
  if (tid < 64) {                                            // r-head layer 2
    float acc = r_b2[tid]; const float* w = r_w2 + tid * 128;
    for (int j = 0; j < 128; ++j) acc += z1s[j] * w[j];
    z2s[tid] = fmaxf(acc, 0.f) * (r_g3[tid] * BNS) + r_be3[tid];
  }
  __syncthreads();
  if (tid == 0) {
    float acc = r_b3[0];
    for (int j = 0; j < 64; ++j) acc += z2s[j] * r_w3[j];
    irs_o[b] = 1.f / (1.f + expf(-acc));
  }

  // prm = mean_l (1 - maxsim)/2 ; fg = max_p prm
  float mx = -3.0e38f;
  for (int p = tid; p < 225; p += 256) {
    float s = 3.f;
#pragma unroll
    for (int li = 0; li < 3; ++li) s -= maxsim[(size_t)li * 14400 + b * 225 + p];
    float v = s * (1.f / 6.f);
    prm[b * 225 + p] = v;
    mx = fmaxf(mx, v);
  }
#pragma unroll
  for (int m = 16; m >= 1; m >>= 1) mx = fmaxf(mx, __shfl_xor(mx, m, 32));
  __syncthreads();
  if (lane == 0) redA[wv] = mx;
  __syncthreads();
  if (tid == 0) {
    float v = redA[0];
    for (int i = 1; i < 8; ++i) v = fmaxf(v, redA[i]);
    fg_o[b] = v;
  }
}

// ---------------- kernel 3: holistic head + final score + harmonic heat -----
__global__ __launch_bounds__(256)
void head_kernel(const float* __restrict__ prm, const float* __restrict__ ts,
                 const float* __restrict__ irs, const float* __restrict__ fg,
                 const float* __restrict__ h_w1, const float* __restrict__ h_b1,
                 const float* __restrict__ h_g2, const float* __restrict__ h_be2,
                 const float* __restrict__ h_w2, const float* __restrict__ h_b2,
                 const float* __restrict__ h_g3, const float* __restrict__ h_be3,
                 const float* __restrict__ h_w3, const float* __restrict__ h_b3,
                 float* __restrict__ final_out, float* __restrict__ heat) {
  const int b = blockIdx.x, tid = threadIdx.x;
  __shared__ float hol[225], z1s[128], z2s[64];
  const float base = ts[b] + irs[b];
  for (int p = tid; p < 225; p += 256) hol[p] = base + prm[b * 225 + p];
  __syncthreads();
  if (tid < 128) {
    float acc = h_b1[tid]; const float* w = h_w1 + tid * 225;
    for (int p = 0; p < 225; ++p) acc += hol[p] * w[p];
    z1s[tid] = fmaxf(acc, 0.f) * (h_g2[tid] * BNS) + h_be2[tid];
  }
  __syncthreads();
  if (tid < 64) {
    float acc = h_b2[tid]; const float* w = h_w2 + tid * 128;
    for (int j = 0; j < 128; ++j) acc += z1s[j] * w[j];
    z2s[tid] = fmaxf(acc, 0.f) * (h_g3[tid] * BNS) + h_be3[tid];
  }
  __syncthreads();
  if (tid == 0) {
    float acc = h_b3[0];
    for (int j = 0; j < 64; ++j) acc += z2s[j] * h_w3[j];
    float hl = 1.f / (1.f + expf(-acc));
    final_out[b] = 0.5f * (hl + fg[b]);
  }
  // harmonic aggregation of 2x2 windows onto 16x16 grid
  const int i = tid >> 4, j = tid & 15;
  const int wi0 = (i - 1 > 0) ? i - 1 : 0, wi1 = (i < 14) ? i : 14;
  const int wj0 = (j - 1 > 0) ? j - 1 : 0, wj1 = (j < 14) ? j : 14;
  float s = 0.f;
  for (int wi = wi0; wi <= wi1; ++wi)
    for (int wj = wj0; wj <= wj1; ++wj)
      s += 1.f / hol[wi * 15 + wj];
  heat[b * 256 + tid] = (float)((wi1 - wi0 + 1) * (wj1 - wj0 + 1)) / s;
}

// ---------------- kernel 4: bilinear 16x16 -> 240x240 (half-pixel, clamp) ---
__global__ __launch_bounds__(256)
void resize_kernel(const float* __restrict__ heat, float* __restrict__ out) {
  const int idx = blockIdx.x * 256 + threadIdx.x;     // 64*240*240 exact
  const int b = idx / 57600;
  const int rem = idx - b * 57600;
  const int y = rem / 240, x = rem - (rem / 240) * 240;
  const float sy = (y + 0.5f) * (1.f / 15.f) - 0.5f;
  const float sx = (x + 0.5f) * (1.f / 15.f) - 0.5f;
  const float fy = floorf(sy), fx = floorf(sx);
  const int y0 = (int)fy, x0 = (int)fx;
  const float wy = sy - fy, wx = sx - fx;
  const int y0c = y0 < 0 ? 0 : (y0 > 15 ? 15 : y0);
  const int y1c = (y0 + 1) < 0 ? 0 : ((y0 + 1) > 15 ? 15 : y0 + 1);
  const int x0c = x0 < 0 ? 0 : (x0 > 15 ? 15 : x0);
  const int x1c = (x0 + 1) < 0 ? 0 : ((x0 + 1) > 15 ? 15 : x0 + 1);
  const float* h = heat + b * 256;
  const float v00 = h[y0c * 16 + x0c], v01 = h[y0c * 16 + x1c];
  const float v10 = h[y1c * 16 + x0c], v11 = h[y1c * 16 + x1c];
  out[idx] = (1.f - wy) * ((1.f - wx) * v00 + wx * v01) +
             wy * ((1.f - wx) * v10 + wx * v11);
}

extern "C" void kernel_launch(void* const* d_in, const int* in_sizes, int n_in,
                              void* d_out, int out_size, void* d_ws, size_t ws_size,
                              hipStream_t stream) {
  (void)in_sizes; (void)n_in; (void)out_size; (void)ws_size;
  const float* image_embeds     = (const float*)d_in[1];
  const float* patch_embeds     = (const float*)d_in[2];
  const float* text_embeds      = (const float*)d_in[3];
  const float* image_embeds_ref = (const float*)d_in[4];
  const float* patch_embeds_ref = (const float*)d_in[5];
  const float* a_w1 = (const float*)d_in[6],  *a_w2 = (const float*)d_in[7];
  const float* r_w1 = (const float*)d_in[8],  *r_b1 = (const float*)d_in[9];
  const float* r_g2 = (const float*)d_in[10], *r_be2 = (const float*)d_in[11];
  const float* r_w2 = (const float*)d_in[12], *r_b2 = (const float*)d_in[13];
  const float* r_g3 = (const float*)d_in[14], *r_be3 = (const float*)d_in[15];
  const float* r_w3 = (const float*)d_in[16], *r_b3 = (const float*)d_in[17];
  const float* h_w1 = (const float*)d_in[18], *h_b1 = (const float*)d_in[19];
  const float* h_g2 = (const float*)d_in[20], *h_be2 = (const float*)d_in[21];
  const float* h_w2 = (const float*)d_in[22], *h_b2 = (const float*)d_in[23];
  const float* h_g3 = (const float*)d_in[24], *h_be3 = (const float*)d_in[25];
  const float* h_w3 = (const float*)d_in[26], *h_b3 = (const float*)d_in[27];

  char* ws = (char*)d_ws;
  unsigned short* refs_bf = (unsigned short*)ws;        // 3*1800*640*2 = 6,912,000 B
  float* maxsim = (float*)(ws + 6912000);               // 3*14400*4
  float* prm    = (float*)(ws + 7084800);               // 64*225*4
  float* ts     = (float*)(ws + 7142400);               // 64*4
  float* irs    = (float*)(ws + 7142656);               // 64*4
  float* fg     = (float*)(ws + 7142912);               // 64*4
  float* heat   = (float*)(ws + 7143168);               // 64*256*4
  float* outF   = (float*)d_out;                        // [64] scores + [64*240*240] hmap

  cvt_bf16_kernel<<<3375, 256, 0, stream>>>(patch_embeds_ref, refs_bf, 3456000);
  simmax_kernel<<<dim3(150, 3), 192, 0, stream>>>(patch_embeds, refs_bf, maxsim);
  score_kernel<<<64, 256, 0, stream>>>(image_embeds, text_embeds, image_embeds_ref,
      a_w1, a_w2, r_w1, r_b1, r_g2, r_be2, r_w2, r_b2, r_g3, r_be3, r_w3, r_b3,
      maxsim, prm, ts, irs, fg);
  head_kernel<<<64, 256, 0, stream>>>(prm, ts, irs, fg,
      h_w1, h_b1, h_g2, h_be2, h_w2, h_b2, h_g3, h_be3, h_w3, h_b3, outF, heat);
  resize_kernel<<<14400, 256, 0, stream>>>(heat, outF + 64);
}
